// Model_12945031430922
// MI455X (gfx1250) — matrix operand order repaired and checked by reading, not compiled, not run on hardware
//
#include <hip/hip_runtime.h>

#define NT 32           // time-tile per workgroup
#define MODE_PLAIN 0
#define MODE_LRELU 1
#define MODE_GLU   2
#define MODE_FINAL 3
#define MODE_VQ    4

typedef __attribute__((ext_vector_type(16))) __bf16 v16bf;
typedef __attribute__((ext_vector_type(8)))  float  v8f;
typedef __attribute__((ext_vector_type(8)))  unsigned int v8u;

// ---- WMMA fragment loaders (wave32 layouts from CDNA5 ISA 7.12.2) ----
// A 16x32 bf16: lane l -> m = m0+(l&15), kh=l>>4; elem j: k = (j<8)? kh*8+j : 16+kh*8+(j-8)
// LDS A tile: [Cout][36] bf16 (stride 36 -> 18 dwords, 8B-aligned rows)
__device__ __forceinline__ v16bf load_a_frag(const __bf16* sA, int m0, int lane) {
  const int m   = m0 + (lane & 15);
  const int kh  = lane >> 4;
  const unsigned* p = (const unsigned*)sA;
  const int row = m * 18;
  v8u u;
#pragma unroll
  for (int i = 0; i < 4; ++i) u[i]     = p[row + kh * 4 + i];
#pragma unroll
  for (int i = 0; i < 4; ++i) u[4 + i] = p[row + 8 + kh * 4 + i];
  return __builtin_bit_cast(v16bf, u);
}

// B 32x16 bf16: lane l -> n = n0+(l&15), kh=l>>4; elem j: k = (j<8)? kh*8+j : 16+kh*8+(j-8)
// LDS B tile stored TRANSPOSED: [NT][36] bf16, sBt[n][k]
__device__ __forceinline__ v16bf load_b_frag(const __bf16* sB, int n0, int lane) {
  const int n  = n0 + (lane & 15);
  const int kh = lane >> 4;
  const unsigned* p = (const unsigned*)sB;
  const int base = n * 18 + kh * 4;
  v8u u;
#pragma unroll
  for (int i = 0; i < 8; ++i) u[i] = p[base + (i & 3) + ((i >> 2) << 3)];
  return __builtin_bit_cast(v16bf, u);
}

__device__ __forceinline__ unsigned orderF(float f) {  // order-preserving float->uint
  unsigned u = __float_as_uint(f);
  return (u & 0x80000000u) ? ~u : (u | 0x80000000u);
}

// ============================================================================
// conv1d-as-implicit-GEMM, fully specialized at compile time:
//   C0/C1  : input channel counts (in1 concatenated after in0)
//   TILES_M: Cout/16
//   MODE   : epilogue (plain / LN+lrelu / LN+GLU / final-f32-transposed / VQ)
//   KS     : kernel width (1 or 5)
// Tile: full Cout x NT time positions per workgroup; 256 threads = 8 waves.
// ============================================================================
template <int C0, int C1, int TILES_M, int MODE, int KS>
__global__ __launch_bounds__(256) void conv_gemm_wmma(
    const __bf16* __restrict__ in0,
    const __bf16* __restrict__ in1,
    const __bf16* __restrict__ W,        // [Cout][Cin*KS] bf16 (VQ: codebook [512][64])
    const float*  __restrict__ bias,
    const float*  __restrict__ gamma,
    const float*  __restrict__ beta,
    __bf16* __restrict__ outb,
    float*  __restrict__ outf,
    int T)
{
  constexpr int Cout  = TILES_M * 16;
  constexpr int Cin   = C0 + C1;
  constexpr int Ktot  = Cin * KS;
  constexpr int NI    = (TILES_M + 3) / 4;           // M-tiles per wave
  constexpr bool RAGGED = (Ktot % 32) != 0;          // e.g. enc0: Ktot = 400
  constexpr bool GUARD  = (TILES_M & 3) != 0;        // e.g. dec2: TILES_M = 5
  constexpr int pad   = (KS - 1) / 2;

  __shared__ __align__(16) __bf16 sA[Cout * 36];     // weight K-chunk
  __shared__ __align__(16) __bf16 sB[NT * 36];       // im2col chunk (transposed)
  __shared__ float sSum[NT], sSumSq[NT], sMean[NT], sRstd[NT];
  __shared__ unsigned long long sKey[NT];

  const int tid  = threadIdx.x;
  const int lane = tid & 31;
  const int wv   = tid >> 5;
  const int b    = blockIdx.y;
  const int t0   = blockIdx.x * NT;
  const int tmb  = wv >> 1;                          // wave's M-tile base
  const int nb   = (wv & 1) << 4;                    // wave's N halftile base

  if (tid < NT) { sSum[tid] = 0.f; sSumSq[tid] = 0.f; sKey[tid] = 0ull; }

  v8f acc[NI];
#pragma unroll
  for (int i = 0; i < NI; ++i)
#pragma unroll
    for (int r = 0; r < 8; ++r) acc[i][r] = 0.f;

#pragma unroll 1
  for (int k0 = 0; k0 < Ktot; k0 += 32) {
    __syncthreads();
    // ---- stage A: Cout x 32 weights, contiguous along flattened (ci,k) ----
#pragma unroll 1
    for (int idx = tid; idx < (Cout << 3); idx += 256) {
      const int m  = idx >> 3;
      const int kk = (idx & 7) << 2;
      unsigned long long v = 0ull;
      if (!RAGGED || (k0 + kk) < Ktot)
        v = *(const unsigned long long*)(W + (size_t)m * Ktot + k0 + kk);
      *(unsigned long long*)(sA + m * 36 + kk) = v;
    }
    // prefetch next weight chunk (global_prefetch_b8)
    if (k0 + 32 < Ktot && tid < Cout)
      __builtin_prefetch(W + (size_t)tid * Ktot + k0 + 32, 0, 1);

    // ---- stage B (transposed im2col): sB[n][kk] = in[ci][t0+n+kq-pad] ----
#pragma unroll 1
    for (int idx = tid; idx < NT * 32; idx += 256) {
      const int kk = idx >> 5;
      const int n  = idx & 31;
      const int kg = k0 + kk;
      __bf16 v = (__bf16)0.f;
      if (!RAGGED || kg < Ktot) {
        int ci, kq;
        if (KS == 1) { ci = kg; kq = 0; }
        else { ci = (int)(((unsigned)kg * 52429u) >> 18); kq = kg - ci * 5; } // /5
        const int tt = t0 + n + kq - pad;
        if (tt >= 0 && tt < T)
          v = (C1 == 0 || ci < C0) ? in0[((size_t)b * C0 + ci) * T + tt]
                                   : in1[((size_t)b * C1 + (ci - C0)) * T + tt];
      }
      sB[n * 36 + kk] = v;
    }
    __syncthreads();

    // ---- WMMA burst: hoist all fragment loads, then NI matrix ops ----
    const v16bf bf = load_b_frag(sB, nb, lane);
    v16bf af[NI];
#pragma unroll
    for (int i = 0; i < NI; ++i) {
      const int tm = tmb + (i << 2);
      if (!GUARD || tm < TILES_M) af[i] = load_a_frag(sA, tm << 4, lane);
    }
#pragma unroll
    for (int i = 0; i < NI; ++i) {
      const int tm = tmb + (i << 2);
      if (!GUARD || tm < TILES_M)
        acc[i] = __builtin_amdgcn_wmma_f32_16x16x32_bf16(
            false, af[i], false, bf, (short)0, acc[i], false, false);
    }
  }

  // ---- epilogue: bias, LN stats via LDS float atomics, activation, store ----
  const int n  = nb + (lane & 15);                   // this lane's time column
  const int kh = lane >> 4;
  const int tg = t0 + n;

  if (bias) {
#pragma unroll
    for (int i = 0; i < NI; ++i) {
      const int tm = tmb + (i << 2);
      if (!GUARD || tm < TILES_M)
#pragma unroll
        for (int r = 0; r < 8; ++r) acc[i][r] += bias[(tm << 4) + r + (kh << 3)];
    }
  }

  if constexpr (MODE == MODE_LRELU || MODE == MODE_GLU) {
    float s = 0.f, s2 = 0.f;
#pragma unroll
    for (int i = 0; i < NI; ++i)
#pragma unroll
      for (int r = 0; r < 8; ++r) { const float v = acc[i][r]; s += v; s2 += v * v; }
    atomicAdd(&sSum[n], s);
    atomicAdd(&sSumSq[n], s2);
    __syncthreads();
    if (tid < NT) {
      constexpr float inv = 1.f / (float)Cout;
      const float mean = sSum[tid] * inv;
      const float var  = sSumSq[tid] * inv - mean * mean;
      sMean[tid] = mean;
      sRstd[tid] = rsqrtf(var + 1e-5f);
    }
    __syncthreads();
    const float mean = sMean[n], rstd = sRstd[n];
    if constexpr (MODE == MODE_LRELU) {
#pragma unroll
      for (int i = 0; i < NI; ++i) {
        const int tm = tmb + (i << 2);
#pragma unroll
        for (int r = 0; r < 8; ++r) {
          const int m = (tm << 4) + r + (kh << 3);
          float v = (acc[i][r] - mean) * rstd * gamma[m] + beta[m];
          v = v > 0.f ? v : 0.01f * v;
          outb[((size_t)b * Cout + m) * T + tg] = (__bf16)v;
        }
      }
    } else {                                         // GLU: TILES_M==32, pair (i, i+4)
      constexpr int Ch = Cout / 2;
#pragma unroll
      for (int i = 0; i < NI / 2; ++i) {
        const int tm = tmb + (i << 2);
#pragma unroll
        for (int r = 0; r < 8; ++r) {
          const int m = (tm << 4) + r + (kh << 3);
          const float a = (acc[i][r]          - mean) * rstd * gamma[m]      + beta[m];
          const float g = (acc[i + NI/2][r]   - mean) * rstd * gamma[m + Ch] + beta[m + Ch];
          outb[((size_t)b * Ch + m) * T + tg] = (__bf16)(a * (1.f / (1.f + __expf(-g))));
        }
      }
    }
  } else if constexpr (MODE == MODE_VQ) {
    // scores[m][n] in regs; per-lane argmax then u64 atomicMax (tie -> smallest idx)
    float best = -3.4e38f; int bi = 0;
#pragma unroll
    for (int i = 0; i < NI; ++i) {
      const int tm = tmb + (i << 2);
#pragma unroll
      for (int r = 0; r < 8; ++r) {
        const int m = (tm << 4) + r + (kh << 3);
        const float v = acc[i][r];
        if (v > best) { best = v; bi = m; }
      }
    }
    const unsigned long long key =
        ((unsigned long long)orderF(best) << 32) | (unsigned)(~bi);
    atomicMax(&sKey[n], key);
    __syncthreads();
    // gather zq[b][d][t] = codebook[best][d]   (C0 == ZD)
    for (int idx = tid; idx < (C0 << 5); idx += 256) {
      const int d = idx >> 5, nn = idx & 31;
      const int sel = (int)(~(unsigned)(sKey[nn] & 0xFFFFFFFFull)) & 0xFFFF;
      outb[((size_t)b * C0 + d) * T + t0 + nn] = W[(size_t)sel * C0 + d];
    }
  } else {                                           // PLAIN bf16 / FINAL f32 transposed
#pragma unroll
    for (int i = 0; i < NI; ++i) {
      const int tm = tmb + (i << 2);
      if (!GUARD || tm < TILES_M)
#pragma unroll
        for (int r = 0; r < 8; ++r) {
          const int m = (tm << 4) + r + (kh << 3);
          const float v = acc[i][r];
          if constexpr (MODE == MODE_FINAL) outf[((size_t)b * T + tg) * Cout + m] = v;
          else                              outb[((size_t)b * Cout + m) * T + tg] = (__bf16)v;
        }
    }
  }
}

// ============================== prep kernels ================================
__global__ void k_cvt(const float* __restrict__ s, __bf16* __restrict__ d, int n) {
  int i = blockIdx.x * 256 + threadIdx.x;
  if (i < n) d[i] = (__bf16)s[i];
}

// deconv weight: src [Cin][Cout][K] -> dst[co][ci][k] = src[ci][co][K-1-k]
__global__ void k_flipT(const float* __restrict__ s, __bf16* __restrict__ d,
                        int Cin, int Cout, int K) {
  int i = blockIdx.x * 256 + threadIdx.x;
  int n = Cin * Cout * K;
  if (i < n) {
    int k = i % K, q = i / K, ci = q % Cin, co = q / Cin;
    d[i] = (__bf16)s[((size_t)ci * Cout + co) * K + (K - 1 - k)];
  }
}

__global__ void k_l2n(const float* __restrict__ s, __bf16* __restrict__ d,
                      int rows, int dim) {
  int r = blockIdx.x * 256 + threadIdx.x;
  if (r < rows) {
    float ss = 0.f;
    for (int j = 0; j < dim; ++j) { float v = s[(size_t)r * dim + j]; ss += v * v; }
    float inv = rsqrtf(ss + 1e-12f);
    for (int j = 0; j < dim; ++j) d[(size_t)r * dim + j] = (__bf16)(s[(size_t)r * dim + j] * inv);
  }
}

__global__ void k_transpose_x(const float* __restrict__ x, __bf16* __restrict__ xt,
                              int Bq, int Tq, int D) {
  size_t i = (size_t)blockIdx.x * 256 + threadIdx.x;
  size_t n = (size_t)Bq * Tq * D;
  if (i < n) {
    int d = (int)(i % D); size_t q = i / D; int t = (int)(q % Tq); int b = (int)(q / Tq);
    xt[((size_t)b * D + d) * Tq + t] = (__bf16)x[i];
  }
}

__global__ void k_gather_y(const int* __restrict__ y, const __bf16* __restrict__ nspk,
                           __bf16* __restrict__ yv, int Bq, int Tq, int YDq) {
  size_t i = (size_t)blockIdx.x * 256 + threadIdx.x;
  size_t n = (size_t)Bq * Tq * YDq;
  if (i < n) {
    int d = (int)(i % YDq); size_t q = i / YDq; int t = (int)(q % Tq); int b = (int)(q / Tq);
    int r = y[(size_t)b * Tq + t];
    yv[((size_t)b * YDq + d) * Tq + t] = nspk[(size_t)r * YDq + d];
  }
}

// ============================== launcher ====================================
extern "C" void kernel_launch(void* const* d_in, const int* in_sizes, int n_in,
                              void* d_out, int out_size, void* d_ws, size_t ws_size,
                              hipStream_t stream) {
  (void)in_sizes; (void)n_in; (void)out_size; (void)ws_size;
  const int Bq = 8, Tq = 16384, DIN = 80, Cq = 256, ZDq = 64, ZNq = 512,
            YDq = 128, Kq = 5;

  const float* x    = (const float*)d_in[0];
  const int*   y    = (const int*)  d_in[1];
  const float* ew0  = (const float*)d_in[2];  const float* eb0 = (const float*)d_in[3];
  const float* eg0  = (const float*)d_in[4];  const float* ebe0= (const float*)d_in[5];
  const float* ew1  = (const float*)d_in[6];  const float* eb1 = (const float*)d_in[7];
  const float* eg1  = (const float*)d_in[8];  const float* ebe1= (const float*)d_in[9];
  const float* ew2  = (const float*)d_in[10]; const float* eb2 = (const float*)d_in[11];
  const float* eg2  = (const float*)d_in[12]; const float* ebe2= (const float*)d_in[13];
  const float* mw   = (const float*)d_in[14]; const float* mb  = (const float*)d_in[15];
  const float* emb  = (const float*)d_in[16];
  const float* spk  = (const float*)d_in[17];
  const float* dw0  = (const float*)d_in[18]; const float* db0 = (const float*)d_in[19];
  const float* dg0  = (const float*)d_in[20]; const float* dbe0= (const float*)d_in[21];
  const float* dw1  = (const float*)d_in[22]; const float* db1 = (const float*)d_in[23];
  const float* dg1  = (const float*)d_in[24]; const float* dbe1= (const float*)d_in[25];
  const float* dw2  = (const float*)d_in[26]; const float* db2 = (const float*)d_in[27];

  char* ws = (char*)d_ws;
  size_t off = 0;
  auto alloc = [&](size_t bytes) -> void* {
    void* p = ws + off;
    off = (off + bytes + 255) & ~(size_t)255;
    return p;
  };

  __bf16* xt   = (__bf16*)alloc((size_t)Bq * DIN * Tq * 2);
  __bf16* bufA = (__bf16*)alloc((size_t)Bq * Cq  * Tq * 2);
  __bf16* bufB = (__bf16*)alloc((size_t)Bq * Cq  * Tq * 2);
  __bf16* yv   = (__bf16*)alloc((size_t)Bq * YDq * Tq * 2);
  __bf16* zb   = (__bf16*)alloc((size_t)Bq * ZDq * Tq * 2);
  __bf16* zq   = (__bf16*)alloc((size_t)Bq * ZDq * Tq * 2);
  __bf16* wE0  = (__bf16*)alloc((size_t)Cq * DIN * Kq * 2);
  __bf16* wE1  = (__bf16*)alloc((size_t)Cq * Cq  * Kq * 2);
  __bf16* wE2  = (__bf16*)alloc((size_t)Cq * Cq  * Kq * 2);
  __bf16* wM   = (__bf16*)alloc((size_t)ZDq * Cq * 2);
  __bf16* wD0  = (__bf16*)alloc((size_t)(2*Cq) * (ZDq + YDq) * Kq * 2);
  __bf16* wD1  = (__bf16*)alloc((size_t)(2*Cq) * (Cq  + YDq) * Kq * 2);
  __bf16* wD2  = (__bf16*)alloc((size_t)DIN    * (Cq  + YDq) * Kq * 2);
  __bf16* nemb = (__bf16*)alloc((size_t)ZNq * ZDq * 2);
  __bf16* nspk = (__bf16*)alloc((size_t)100 * YDq * 2);

  auto blk = [](size_t n) { return dim3((unsigned)((n + 255) / 256)); };

  // ---- weight / embedding prep ----
  k_cvt<<<blk(Cq*DIN*Kq), 256, 0, stream>>>(ew0, wE0, Cq*DIN*Kq);
  k_cvt<<<blk(Cq*Cq*Kq),  256, 0, stream>>>(ew1, wE1, Cq*Cq*Kq);
  k_cvt<<<blk(Cq*Cq*Kq),  256, 0, stream>>>(ew2, wE2, Cq*Cq*Kq);
  k_cvt<<<blk(ZDq*Cq),    256, 0, stream>>>(mw,  wM,  ZDq*Cq);
  k_flipT<<<blk((size_t)(ZDq+YDq)*2*Cq*Kq), 256, 0, stream>>>(dw0, wD0, ZDq+YDq, 2*Cq, Kq);
  k_flipT<<<blk((size_t)(Cq +YDq)*2*Cq*Kq), 256, 0, stream>>>(dw1, wD1, Cq +YDq, 2*Cq, Kq);
  k_flipT<<<blk((size_t)(Cq +YDq)*DIN*Kq),  256, 0, stream>>>(dw2, wD2, Cq +YDq, DIN,  Kq);
  k_l2n<<<blk(ZNq), 256, 0, stream>>>(emb, nemb, ZNq, ZDq);
  k_l2n<<<blk(100), 256, 0, stream>>>(spk, nspk, 100, YDq);
  k_transpose_x<<<blk((size_t)Bq*Tq*DIN), 256, 0, stream>>>(x, xt, Bq, Tq, DIN);
  k_gather_y<<<blk((size_t)Bq*Tq*YDq), 256, 0, stream>>>(y, nspk, yv, Bq, Tq, YDq);

  // ---- pipeline ----
  const dim3 grid(Tq / NT, Bq);
  const __bf16* nb = nullptr;
  const float*  nf = nullptr;
  // encoder: conv(K=5) + LN + leaky-relu
  conv_gemm_wmma< 80,   0, 16, MODE_LRELU, 5><<<grid, 256, 0, stream>>>(
      xt,   nb, wE0, eb0, eg0, ebe0, bufA, nullptr, Tq);
  conv_gemm_wmma<256,   0, 16, MODE_LRELU, 5><<<grid, 256, 0, stream>>>(
      bufA, nb, wE1, eb1, eg1, ebe1, bufB, nullptr, Tq);
  conv_gemm_wmma<256,   0, 16, MODE_LRELU, 5><<<grid, 256, 0, stream>>>(
      bufB, nb, wE2, eb2, eg2, ebe2, bufA, nullptr, Tq);
  // 1x1 projection to z
  conv_gemm_wmma<256,   0,  4, MODE_PLAIN, 1><<<grid, 256, 0, stream>>>(
      bufA, nb, wM,  mb,  nf,  nf,   zb,   nullptr, Tq);
  // VQ: scores = codebook(512x64) @ z, argmax, gather (zq == q in fwd)
  conv_gemm_wmma< 64,   0, 32, MODE_VQ,    1><<<grid, 256, 0, stream>>>(
      zb,   nb, nemb, nf, nf,  nf,   zq,   nullptr, Tq);
  // decoder: deconv(K=5) + LN + GLU
  conv_gemm_wmma< 64, 128, 32, MODE_GLU,   5><<<grid, 256, 0, stream>>>(
      zq,   yv, wD0, db0, dg0, dbe0, bufA, nullptr, Tq);
  conv_gemm_wmma<256, 128, 32, MODE_GLU,   5><<<grid, 256, 0, stream>>>(
      bufA, yv, wD1, db1, dg1, dbe1, bufB, nullptr, Tq);
  // final ConvTranspose -> d_out [B,T,80] fp32 (written transposed)
  conv_gemm_wmma<256, 128,  5, MODE_FINAL, 5><<<grid, 256, 0, stream>>>(
      bufB, yv, wD2, db2, nf,  nf,   (__bf16*)nullptr, (float*)d_out, Tq);
}
